// DeepMHCII_EL_Split_AttMIL_59425167507651
// MI455X (gfx1250) — matrix operand order repaired and checked by reading, compiled, not run on hardware
//
#include <hip/hip_runtime.h>
#include <hip/hip_bf16.h>
#include <math.h>

typedef _Float16 f16;
typedef __attribute__((ext_vector_type(16))) _Float16 v16h;
typedef __attribute__((ext_vector_type(8)))  _Float16 v8h;
typedef __attribute__((ext_vector_type(8)))  float    v8f;

#define NN 1024
#define LL 27
#define MM 34
#define EE 16
#define BN_EPS 1e-5f

// ---------------- workspace layout (bytes) ----------------
#define O_PEP16   0ull                 // N*27*16 f16            = 884736
#define O_MHC16   884736ull            // N*34*16 f16            = 1114112
#define O_WCAT    1998848ull           // 6912*64 f16            = 884736
#define O_WLIN    2883584ull           // 917504 f16             = 1835008
#define O_CB      4718592ull           // 1280 f32
#define O_CS      4723712ull
#define O_CBE     4728832ull
#define O_POOL    4733952ull           // N*256*3 f32            = 3145728
#define O_XF      7879680ull           // 21504*640 f16          = 27525120
#define O_XR      35404800ull          // 43008*640 f16          = 55050240
#define O_H       90455040ull          // 43008*512 f16          = 44040192
// total = 134495232 bytes

struct ConvPtrs {
    const float* w[6];
    const float* b[6];
    const float* g[6];
    const float* be[6];
};

// ---------------- positional encoding ----------------
__device__ __forceinline__ float pe_val(int pos, int e) {
    int j = e >> 1;
    float div = expf(-(float)(2 * j) * logf(10000.f) / 16.f);
    float t = (float)pos * div;
    return (e & 1) ? cosf(t) : sinf(t);
}

// ---------------- prep: embeddings (f16) ----------------
__global__ void k_embed(const int* __restrict__ pep_x, const int* __restrict__ mhc_x,
                        const float* __restrict__ emb_pep, const float* __restrict__ emb_mhc,
                        f16* __restrict__ pep16, f16* __restrict__ mhc16) {
    int idx = blockIdx.x * blockDim.x + threadIdx.x;
    const int pepN = NN * LL * EE;
    const int mhcN = NN * MM * EE;
    if (idx < pepN) {
        int n = idx / (LL * EE), r = idx % (LL * EE), p = r / EE, e = r % EE;
        float v = emb_pep[pep_x[n * LL + p] * EE + e];
        if (p >= 3 && p < 24) v += pe_val(p - 3, e);
        pep16[idx] = (f16)v;
    } else if (idx < pepN + mhcN) {
        int k = idx - pepN;
        int n = k / (MM * EE), r = k % (MM * EE), m = r / EE, e = r % EE;
        float v = emb_mhc[mhc_x[n * MM + m] * EE + e] + pe_val(m, e);
        mhc16[k] = (f16)v;
    }
}

// ---------------- prep: conv weights (K-padded to 64) + BN packs ----------------
__global__ void k_wconv(ConvPtrs P, f16* __restrict__ wcat,
                        float* __restrict__ cB, float* __restrict__ cS, float* __restrict__ cBe) {
    const int kd_[6]   = {3, 5, 7, 3, 5, 7};
    const int base_[6] = {0, 384, 1664, 3456, 3840, 5120};
    int idx = blockIdx.x * blockDim.x + threadIdx.x;
    if (idx < 6912 * 64) {
        int row = idx >> 6, m = idx & 63;
        int p = 0;
        while (p < 5 && row >= base_[p + 1]) ++p;
        int local = row - base_[p];
        int kd = kd_[p];
        int c = local / kd, k = local % kd;
        float v = (m < MM) ? P.w[p][(c * kd + k) * MM + m] : 0.f;
        wcat[idx] = (f16)v;
    } else if (idx < 6912 * 64 + 1280) {
        int ch = idx - 6912 * 64;
        int br = ch / 640, cc = ch % 640;
        int j, c;
        if (cc < 128)      { j = 0; c = cc; }
        else if (cc < 384) { j = 1; c = cc - 128; }
        else               { j = 2; c = cc - 384; }
        int p = br * 3 + j;
        cB[ch]  = P.b[p][c];
        cS[ch]  = P.g[p][c] * rsqrtf(1.f + BN_EPS);
        cBe[ch] = P.be[p][c];
    }
}

// ---------------- prep: linear weights -> f16 pack ----------------
__global__ void k_wlin(const float* __restrict__ w0, const float* __restrict__ w1,
                       const float* __restrict__ w2, const float* __restrict__ w3,
                       f16* __restrict__ dst) {
    int idx = blockIdx.x * blockDim.x + threadIdx.x;
    if (idx < 327680)       dst[idx] = (f16)w0[idx];
    else if (idx < 458752)  dst[idx] = (f16)w1[idx - 327680];
    else if (idx < 786432)  dst[idx] = (f16)w2[idx - 458752];
    else if (idx < 917504)  dst[idx] = (f16)w3[idx - 786432];
}

// ---------------- prep: zero pool ----------------
__global__ void k_zero(int* __restrict__ p, int n) {
    int idx = blockIdx.x * blockDim.x + threadIdx.x;
    if (idx < n) p[idx] = 0;
}

// ---------------- fused conv: WMMA Kb stage + WMMA conv stage, BN+ReLU ----------------
__global__ __launch_bounds__(256) void k_conv(
    const f16* __restrict__ wcat, const f16* __restrict__ pep16, const f16* __restrict__ mhc16,
    const float* __restrict__ cB, const float* __restrict__ cS, const float* __restrict__ cBe,
    f16* __restrict__ Xf, f16* __restrict__ Xr) {
    __shared__ __align__(32) f16 mhcL[64 * 16];
    __shared__ __align__(32) f16 pepL[LL * 16];
    __shared__ __align__(32) f16 KbL[1792 * 16];

    const int kdArr[6]    = {3, 5, 7, 3, 5, 7};
    const int cArr[6]     = {128, 256, 256, 128, 256, 256};
    const int offArr[6]   = {2, 1, 0, 2, 1, 0};
    const int baseArr[6]  = {0, 384, 1664, 3456, 3840, 5120};
    const int cbaseArr[6] = {0, 128, 384, 0, 128, 384};

    const int pair = blockIdx.y;
    const int n    = blockIdx.x;
    const int kd = kdArr[pair], cN = cArr[pair], off = offArr[pair];
    const int rbase = baseArr[pair], cbase = cbaseArr[pair];
    const int tid = threadIdx.x;

    for (int t = tid; t < 64 * 16; t += 256)
        mhcL[t] = (t < MM * 16) ? mhc16[(size_t)n * MM * 16 + t] : (f16)0;
    for (int t = tid; t < LL * 16; t += 256)
        pepL[t] = pep16[(size_t)n * LL * 16 + t];
    __syncthreads();

    const int lane = tid & 31, wave = tid >> 5;
    const int col = lane & 15, hi = lane >> 4;

    // B fragments for stage 1, K chunks 0..31 and 32..63 (mhc is KxN = 34x16, zero-padded)
    v16h b0, b1;
#pragma unroll
    for (int j = 0; j < 16; ++j) {
        int k0 = hi * 8 + j + (j >= 8 ? 8 : 0);
        int k1 = 32 + hi * 8 + j + (j >= 8 ? 8 : 0);
        b0[j] = (k0 < MM) ? mhcL[k0 * 16 + col] : (f16)0;
        b1[j] = (k1 < MM) ? mhcL[k1 * 16 + col] : (f16)0;
    }

    // stage 1: Kb(rows x 16) = Wcat(rows x 64) @ mhc(64 x 16) via WMMA
    const int nTiles = (cN * kd) >> 4;
    for (int t = wave; t < nTiles; t += 8) {
        int rowG = rbase + t * 16 + col;
        const f16* ap = wcat + (size_t)rowG * 64;
        v8h a0lo = *(const v8h*)(ap + (hi ? 8 : 0));
        v8h a0hi = *(const v8h*)(ap + 16 + (hi ? 8 : 0));
        v8h a1lo = *(const v8h*)(ap + 32 + (hi ? 8 : 0));
        v8h a1hi = *(const v8h*)(ap + 48 + (hi ? 8 : 0));
        v16h a0, a1;
#pragma unroll
        for (int j = 0; j < 8; ++j) {
            a0[j] = a0lo[j]; a0[j + 8] = a0hi[j];
            a1[j] = a1lo[j]; a1[j + 8] = a1hi[j];
        }
        v8f acc = {};
        acc = __builtin_amdgcn_wmma_f32_16x16x32_f16(false, a0, false, b0, (short)0, acc, false, false);
        acc = __builtin_amdgcn_wmma_f32_16x16x32_f16(false, a1, false, b1, (short)0, acc, false, false);
#pragma unroll
        for (int v = 0; v < 8; ++v) {
            int rr = t * 16 + v + (hi ? 8 : 0);
            KbL[rr * 16 + col] = (f16)acc[v];
        }
    }
    __syncthreads();

    // stage 2 via WMMA: X(c x cols) = Kb(c x kd*16) @ P(kd*16 x cols)
    // P[(k,e), col] gathers peptide rows (with r-branch reversals folded in).
    const int kd16  = kd * 16;
    const int nCh   = (kd16 + 31) >> 5;           // 2,3,4 (uniform per block)
    const int nNt   = (pair < 3) ? 2 : 3;
    const int nMt   = cN >> 4;                    // 8 or 16 (multiple of 8)
    const int colLim = (pair < 3) ? 21 : 42;

    for (int nt2 = 0; nt2 < nNt; ++nt2) {
        int colg = nt2 * 16 + col;
        bool colOk = colg < colLim;
        // build B fragments for this column tile (reused across all M tiles)
        v16h bfr[4];
#pragma unroll
        for (int ch = 0; ch < 4; ++ch) {
            v16h bb;
#pragma unroll
            for (int j = 0; j < 16; ++j) {
                int kk = ch * 32 + (hi ? 8 : 0) + j + (j >= 8 ? 8 : 0);
                f16 v = (f16)0;
                if (ch < nCh && kk < kd16 && colOk) {
                    int k = kk >> 4, e = kk & 15;
                    int prow;
                    if (pair < 3) {
                        prow = off + k + colg;
                    } else {
                        int w = colg / 21, i = colg % 21;
                        int pos = off + (kd - 1 - k) + i;      // kernel reversal
                        prow = (w == 0) ? (LL - 1 - pos) : pos; // pep_i w=0 reversed
                    }
                    v = pepL[prow * 16 + e];
                }
                bb[j] = v;
            }
            bfr[ch] = bb;
        }
        for (int mt = wave; mt < nMt; mt += 8) {
            int rowc = mt * 16 + col;   // c index for A row (lane&15)
            v8f acc = {};
#pragma unroll
            for (int ch = 0; ch < 4; ++ch) {
                if (ch < nCh) {                       // uniform branch
                    int g0 = ch * 32 + (hi ? 8 : 0);
                    int g1 = g0 + 16;
                    v8h l = {};
                    v8h h = {};
                    if (g0 < kd16) l = *(const v8h*)(&KbL[rowc * kd16 + g0]);
                    if (g1 < kd16) h = *(const v8h*)(&KbL[rowc * kd16 + g1]);
                    v16h a;
#pragma unroll
                    for (int j = 0; j < 8; ++j) { a[j] = l[j]; a[j + 8] = h[j]; }
                    acc = __builtin_amdgcn_wmma_f32_16x16x32_f16(false, a, false, bfr[ch],
                                                                 (short)0, acc, false, false);
                }
            }
            // epilogue: BN + ReLU, scatter to GEMM-ready activation layouts
            if (colOk) {
#pragma unroll
                for (int v = 0; v < 8; ++v) {
                    int cloc = mt * 16 + v + (hi ? 8 : 0);
                    int chg = (pair < 3 ? 0 : 640) + cbase + cloc;
                    float y = fmaxf(0.f, (acc[v] + cB[chg]) * cS[chg] + cBe[chg]);
                    if (pair < 3) {
                        Xf[(size_t)(n * 21 + colg) * 640 + cbase + cloc] = (f16)y;
                    } else {
                        int w = colg / 21, i = colg % 21;
                        int iout = (w == 0) ? (20 - i) : i;    // conv_r i-reversal for w=0
                        Xr[(size_t)(n * 42 + w * 21 + iout) * 640 + (cbase + cloc)] = (f16)y;
                    }
                }
            }
        }
    }
}

// ---------------- WMMA GEMM: C = ReLU(BN(A @ W^T + b)); optional fused max-pool ----------------
__device__ __forceinline__ v16h load_afrag(const f16* ap) {
    v8h l = *(const v8h*)ap;
    v8h h = *(const v8h*)(ap + 16);
    v16h a;
#pragma unroll
    for (int j = 0; j < 8; ++j) { a[j] = l[j]; a[j + 8] = h[j]; }
    return a;
}

__global__ __launch_bounds__(256) void k_gemm(
    const f16* __restrict__ A, const f16* __restrict__ W,
    const float* __restrict__ bias, const float* __restrict__ gg, const float* __restrict__ be,
    int Mrows, int K, int Nout, int mode,
    f16* __restrict__ out, int* __restrict__ pool, const int* __restrict__ pep_x,
    int wavesTotal) {
    int wid = blockIdx.x * 8 + (threadIdx.x >> 5);
    if (wid >= wavesTotal) return;                 // whole-wave uniform exit
    int lane = threadIdx.x & 31, col16 = lane & 15, hi = lane >> 4;
    int mtiles4 = Mrows >> 6;
    int m64 = wid % mtiles4, nt = wid / mtiles4;
    int col = nt * 16 + col16;
    int rbase = m64 * 64 + col16;

    v8f acc0 = {}, acc1 = {}, acc2 = {}, acc3 = {};
    const f16* wrow = W + (size_t)col * K + (hi ? 8 : 0);
    const size_t hioff = hi ? 8 : 0;
    for (int k0 = 0; k0 < K; k0 += 32) {
        if (k0 + 32 < K) {
            __builtin_prefetch(wrow + k0 + 32, 0, 0);                    // global_prefetch_b8
            __builtin_prefetch(A + (size_t)rbase * K + k0 + 32, 0, 0);
        }
        v16h b = load_afrag(wrow + k0);
        v16h a0 = load_afrag(A + (size_t)(rbase +  0) * K + k0 + hioff);
        v16h a1 = load_afrag(A + (size_t)(rbase + 16) * K + k0 + hioff);
        v16h a2 = load_afrag(A + (size_t)(rbase + 32) * K + k0 + hioff);
        v16h a3 = load_afrag(A + (size_t)(rbase + 48) * K + k0 + hioff);
        acc0 = __builtin_amdgcn_wmma_f32_16x16x32_f16(false, a0, false, b, (short)0, acc0, false, false);
        acc1 = __builtin_amdgcn_wmma_f32_16x16x32_f16(false, a1, false, b, (short)0, acc1, false, false);
        acc2 = __builtin_amdgcn_wmma_f32_16x16x32_f16(false, a2, false, b, (short)0, acc2, false, false);
        acc3 = __builtin_amdgcn_wmma_f32_16x16x32_f16(false, a3, false, b, (short)0, acc3, false, false);
    }

    float bcol = bias[col], scol = gg[col] * rsqrtf(1.f + BN_EPS), ecol = be[col];
#pragma unroll
    for (int t = 0; t < 4; ++t) {
        v8f acc = (t == 0) ? acc0 : (t == 1) ? acc1 : (t == 2) ? acc2 : acc3;
#pragma unroll
        for (int v = 0; v < 8; ++v) {
            int row = m64 * 64 + t * 16 + v + (hi ? 8 : 0);
            float y = fmaxf(0.f, (acc[v] + bcol) * scol + ecol);
            if (mode == 0) {
                out[(size_t)row * Nout + col] = (f16)y;
            } else if (mode == 1) {
                int n = row / 21, i = row % 21;
                if (pep_x[n * LL + 3 + i] != 0)
                    atomicMax(&pool[(n * 256 + col) * 3 + 0], __float_as_int(y));
            } else {
                int n = row / 42, r2 = row % 42, si = r2 / 21, i = r2 % 21;
                if (pep_x[n * LL + 3 + i] != 0)
                    atomicMax(&pool[(n * 256 + col) * 3 + 1 + si], __float_as_int(y));
            }
        }
    }
}

// ---------------- attention + bag output ----------------
__global__ __launch_bounds__(256) void k_final(
    const float* __restrict__ pool, const int* __restrict__ bags, const int* __restrict__ inv,
    const float* __restrict__ wA1, const float* __restrict__ bA1,
    const float* __restrict__ wA2, const float* __restrict__ bA2,
    const float* __restrict__ wOut, const float* __restrict__ bOut,
    float* __restrict__ outp) {
    __shared__ float pl[768];
    __shared__ float feat[256];
    __shared__ float red[256];
    __shared__ float aArr[1024];
    __shared__ float s0Arr[1024];
    __shared__ float s1Arr[1024];
    __shared__ float scv[6];
    __shared__ int ssh[2];

    int b = blockIdx.x, tid = threadIdx.x;
    if (tid == 0) {
        int st = 0;
        for (int j = 0; j < b; ++j) st += bags[j];
        ssh[0] = st; ssh[1] = bags[b];
    }
    __syncthreads();
    int start = ssh[0];
    int cnt = ssh[1]; if (cnt > 1024) cnt = 1024;
    int inverse = inv[0];

    for (int li = 0; li < cnt; ++li) {
        int n = start + li;
        for (int t = tid; t < 768; t += 256) pl[t] = pool[(size_t)n * 768 + t];
        __syncthreads();
        feat[tid] = (pl[tid * 3] + pl[tid * 3 + 1] + pl[tid * 3 + 2]) * (1.f / 3.f);
        __syncthreads();
        float dot = 0.f;
        for (int c = 0; c < 256; ++c) dot += feat[c] * wA1[tid * 256 + c];
        float th = tanhf(dot + bA1[tid]);
        red[tid] = th * wA2[tid];
        __syncthreads();
        for (int s = 128; s > 0; s >>= 1) { if (tid < s) red[tid] += red[tid + s]; __syncthreads(); }
        if (tid == 0) aArr[li] = red[0] + bA2[0];
        __syncthreads();
        for (int ow = 0; ow < 6; ++ow) {
            int o = ow / 3, w = ow % 3;
            red[tid] = wOut[o * 256 + tid] * pl[tid * 3 + w];
            __syncthreads();
            for (int s = 128; s > 0; s >>= 1) { if (tid < s) red[tid] += red[tid + s]; __syncthreads(); }
            if (tid == 0) scv[ow] = red[0];
            __syncthreads();
        }
        if (tid == 0) {
            float sc[2][3];
            for (int o = 0; o < 2; ++o)
                for (int w = 0; w < 3; ++w)
                    sc[o][w] = 1.f / (1.f + expf(-(scv[o * 3 + w] + bOut[o])));
            for (int o = 0; o < 2; ++o) {
                float half = 0.5f * (sc[o][0] + sc[o][1]);
                float sv = inverse ? fmaxf(half, sc[o][2]) : half;
                if (o == 0) s0Arr[li] = sv; else s1Arr[li] = sv;
            }
        }
        __syncthreads();
    }

    // segment softmax + weighted sum
    float mx = -1e30f;
    for (int t = tid; t < cnt; t += 256) mx = fmaxf(mx, aArr[t]);
    red[tid] = mx; __syncthreads();
    for (int s = 128; s > 0; s >>= 1) { if (tid < s) red[tid] = fmaxf(red[tid], red[tid + s]); __syncthreads(); }
    float amax = red[0]; __syncthreads();
    float sm = 0.f;
    for (int t = tid; t < cnt; t += 256) sm += expf(aArr[t] - amax);
    red[tid] = sm; __syncthreads();
    for (int s = 128; s > 0; s >>= 1) { if (tid < s) red[tid] += red[tid + s]; __syncthreads(); }
    float denom = red[0]; __syncthreads();
    for (int o = 0; o < 2; ++o) {
        float acc = 0.f;
        for (int t = tid; t < cnt; t += 256) {
            float att = expf(aArr[t] - amax) / denom;
            acc += att * (o == 0 ? s0Arr[t] : s1Arr[t]);
        }
        red[tid] = acc; __syncthreads();
        for (int s = 128; s > 0; s >>= 1) { if (tid < s) red[tid] += red[tid + s]; __syncthreads(); }
        if (tid == 0) outp[b * 2 + o] = fminf(1.f, fmaxf(0.f, red[0]));
        __syncthreads();
    }
}

// ---------------- launch ----------------
extern "C" void kernel_launch(void* const* d_in, const int* in_sizes, int n_in,
                              void* d_out, int out_size, void* d_ws, size_t ws_size,
                              hipStream_t stream) {
    (void)n_in; (void)out_size; (void)ws_size;
    const int*   pep_x   = (const int*)d_in[0];
    const int*   mhc_x   = (const int*)d_in[1];
    const int*   bags    = (const int*)d_in[4];
    const int*   inv     = (const int*)d_in[5];
    const float* emb_pep = (const float*)d_in[6];
    const float* emb_mhc = (const float*)d_in[7];

    ConvPtrs cp;
    for (int p = 0; p < 6; ++p) {
        cp.w[p]  = (const float*)d_in[8 + p * 4 + 0];
        cp.b[p]  = (const float*)d_in[8 + p * 4 + 1];
        cp.g[p]  = (const float*)d_in[8 + p * 4 + 2];
        cp.be[p] = (const float*)d_in[8 + p * 4 + 3];
    }
    const float* wlf0 = (const float*)d_in[32];
    const float* blf0 = (const float*)d_in[33];
    const float* glf0 = (const float*)d_in[34];
    const float* elf0 = (const float*)d_in[35];
    const float* wlf1 = (const float*)d_in[36];
    const float* blf1 = (const float*)d_in[37];
    const float* glf1 = (const float*)d_in[38];
    const float* elf1 = (const float*)d_in[39];
    const float* wlr0 = (const float*)d_in[40];
    const float* blr0 = (const float*)d_in[41];
    const float* glr0 = (const float*)d_in[42];
    const float* elr0 = (const float*)d_in[43];
    const float* wlr1 = (const float*)d_in[44];
    const float* blr1 = (const float*)d_in[45];
    const float* glr1 = (const float*)d_in[46];
    const float* elr1 = (const float*)d_in[47];
    const float* wA1  = (const float*)d_in[48];
    const float* bA1  = (const float*)d_in[49];
    const float* wA2  = (const float*)d_in[50];
    const float* bA2  = (const float*)d_in[51];
    const float* wOut = (const float*)d_in[52];
    const float* bOut = (const float*)d_in[53];

    char* ws = (char*)d_ws;
    f16*   pep16 = (f16*)(ws + O_PEP16);
    f16*   mhc16 = (f16*)(ws + O_MHC16);
    f16*   wcat  = (f16*)(ws + O_WCAT);
    f16*   wlin  = (f16*)(ws + O_WLIN);
    float* cB    = (float*)(ws + O_CB);
    float* cS    = (float*)(ws + O_CS);
    float* cBe   = (float*)(ws + O_CBE);
    float* pool  = (float*)(ws + O_POOL);
    f16*   Xf    = (f16*)(ws + O_XF);
    f16*   Xr    = (f16*)(ws + O_XR);
    f16*   H     = (f16*)(ws + O_H);

    f16* w16_lf0 = wlin;
    f16* w16_lf1 = wlin + 327680;
    f16* w16_lr0 = wlin + 458752;
    f16* w16_lr1 = wlin + 786432;

    // prep
    {
        int tot = NN * (LL + MM) * EE;
        k_embed<<<(tot + 255) / 256, 256, 0, stream>>>(pep_x, mhc_x, emb_pep, emb_mhc, pep16, mhc16);
    }
    {
        int tot = 6912 * 64 + 1280;
        k_wconv<<<(tot + 255) / 256, 256, 0, stream>>>(cp, wcat, cB, cS, cBe);
    }
    k_wlin<<<(917504 + 255) / 256, 256, 0, stream>>>(wlf0, wlf1, wlr0, wlr1, wlin);
    k_zero<<<(786432 + 255) / 256, 256, 0, stream>>>((int*)pool, 786432);

    // fused conv (WMMA stage-1 + WMMA stage-2)
    k_conv<<<dim3(NN, 6), 256, 0, stream>>>(wcat, pep16, mhc16, cB, cS, cBe, Xf, Xr);

    // MLPs (WMMA GEMMs); last layer of each branch fuses masked max-pool
    const int Mf = NN * 21, Mr = NN * 42;
    {
        int wt = (Mf / 64) * (512 / 16);
        k_gemm<<<(wt + 7) / 8, 256, 0, stream>>>(Xf, w16_lf0, blf0, glf0, elf0,
                                                 Mf, 640, 512, 0, H, (int*)pool, pep_x, wt);
    }
    {
        int wt = (Mf / 64) * (256 / 16);
        k_gemm<<<(wt + 7) / 8, 256, 0, stream>>>(H, w16_lf1, blf1, glf1, elf1,
                                                 Mf, 512, 256, 1, (f16*)nullptr, (int*)pool, pep_x, wt);
    }
    {
        int wt = (Mr / 64) * (512 / 16);
        k_gemm<<<(wt + 7) / 8, 256, 0, stream>>>(Xr, w16_lr0, blr0, glr0, elr0,
                                                 Mr, 640, 512, 0, H, (int*)pool, pep_x, wt);
    }
    {
        int wt = (Mr / 64) * (256 / 16);
        k_gemm<<<(wt + 7) / 8, 256, 0, stream>>>(H, w16_lr1, blr1, glr1, elr1,
                                                 Mr, 512, 256, 2, (f16*)nullptr, (int*)pool, pep_x, wt);
    }

    // attention + output
    int nBags = in_sizes[4];
    k_final<<<nBags, 256, 0, stream>>>(pool, bags, inv, wA1, bA1, wA2, bA2, wOut, bOut, (float*)d_out);
}
